// BaseWindowAttention_17703855194433
// MI455X (gfx1250) — matrix operand order repaired and checked by reading, not compiled, not run on hardware
//
#include <hip/hip_runtime.h>
#include <hip/hip_bf16.h>

// CDNA5 (gfx1250) wave32 WMMA implementation of window attention.
// D = A(16x32 bf16) x B(32x16 bf16) + C(16x16 f32) via v_wmma_f32_16x16x32_bf16.

typedef __attribute__((ext_vector_type(16))) __bf16 bf16x16;
typedef __attribute__((ext_vector_type(8)))  __bf16 bf16x8;
typedef __attribute__((ext_vector_type(8)))  float  f32x8;

#define TOKENS   131072      // 2*4*128*128
#define QKVN     768
#define DIMC     256
#define SCALE    0.17677669529663687f

__device__ __forceinline__ __bf16 f2bf(float f) {
    union { float f; unsigned u; } a; a.f = f;
    unsigned r = a.u + 0x7FFFu + ((a.u >> 16) & 1u);   // round-to-nearest-even
    unsigned short h = (unsigned short)(r >> 16);
    return __builtin_bit_cast(__bf16, h);
}

__device__ __forceinline__ f32x8 wmma_bf16(bf16x16 a, bf16x16 b, f32x8 c) {
    return __builtin_amdgcn_wmma_f32_16x16x32_bf16(false, a, false, b, (short)0, c, false, false);
}

// ---------------------------------------------------------------------------
// Kernel 0a: transpose-convert weights to bf16 (column-major: WT[n][k]).
// Makes every WMMA B-fragment load a single contiguous 32B load per lane.
// ---------------------------------------------------------------------------
__global__ void k_convert_w(const float* __restrict__ wqkv,
                            const float* __restrict__ wout,
                            __bf16* __restrict__ wqT,   // 768 x 256
                            __bf16* __restrict__ woT) { // 256 x 256
    int idx = blockIdx.x * 256 + threadIdx.x;
    if (idx < 768 * 256) {
        int n = idx >> 8, k = idx & 255;
        wqT[n * 256 + k] = f2bf(wqkv[k * 768 + n]);
    } else {
        int j = idx - 768 * 256;     // < 65536
        int n = j >> 8, k = j & 255;
        woT[n * 256 + k] = f2bf(wout[k * 256 + n]);
    }
}

// ---------------------------------------------------------------------------
// Kernel 0b: one-shot x(f32) -> xb(bf16). Hoists conversion VALU out of the
// QKV GEMM inner loop (A rows are reused by 48 N-tiles; convert once, not 48x).
// 4 elements per thread: float4 in, 8B packed bf16 out.
// ---------------------------------------------------------------------------
__global__ void __launch_bounds__(256) k_convert_x(const float* __restrict__ x,
                                                   __bf16* __restrict__ xb) {
    size_t i = ((size_t)blockIdx.x * 256 + threadIdx.x) * 4;   // 33554432 elems
    float4 f = *(const float4*)(x + i);
    union { __bf16 b[4]; uint2 u; } o;
    o.b[0] = f2bf(f.x); o.b[1] = f2bf(f.y); o.b[2] = f2bf(f.z); o.b[3] = f2bf(f.w);
    *(uint2*)(xb + i) = o.u;
}

// ---------------------------------------------------------------------------
// Kernel 1: QKV projection GEMM  qkv(131072x768,bf16) = xb(131072x256,bf16) * WqkvT
// One 16x16 output tile per wave; K=256 -> 8 WMMAs. Inner loop = loads + WMMA only.
// ---------------------------------------------------------------------------
__global__ void __launch_bounds__(256) k_qkv(const __bf16* __restrict__ xb,
                                             const __bf16* __restrict__ wqT,
                                             __bf16*       __restrict__ qkv) {
    int wave = threadIdx.x >> 5, lane = threadIdx.x & 31;
    int t  = blockIdx.x * 8 + wave;          // 393216 tiles total
    int tM = t / 48, tN = t - tM * 48;       // waves in a block share tM -> A reuse in cache
    int ln = lane & 15, hi = lane >> 4;

    const __bf16* arow = xb  + (size_t)(tM * 16 + ln) * DIMC;   // A: row = lane%16
    const __bf16* brow = wqT + (size_t)(tN * 16 + ln) * DIMC;   // B: col = lane%16

    f32x8 acc = {};
#pragma unroll
    for (int k0 = 0; k0 < DIMC; k0 += 32) {
        // A fragment: K = k0 + hi*16 + e, contiguous (matches B)
        union { bf16x16 v; bf16x8 h[2]; } a;
        a.h[0] = *(const bf16x8*)(arow + k0 + hi * 16);
        a.h[1] = *(const bf16x8*)(arow + k0 + hi * 16 + 8);
        // B fragment: K = k0 + hi*16 + e, contiguous in transposed weights
        bf16x16 b = *(const bf16x16*)(brow + k0 + hi * 16);
        acc = wmma_bf16(a.v, b, acc);
    }
    // D layout: col = tN*16 + lane%16, row = tM*16 + r + 8*(lane/16)
    int col = tN * 16 + ln;
#pragma unroll
    for (int r = 0; r < 8; r++) {
        int row = tM * 16 + r + 8 * hi;
        qkv[(size_t)row * QKVN + col] = f2bf(acc[r]);
    }
}

// ---------------------------------------------------------------------------
// Kernel 2: window attention. One wave per (window, head): 64x64x32 attention.
// 16 WMMAs for S = Q K^T, fp32 softmax (+rel-pos bias), 16 WMMAs for P V.
// ---------------------------------------------------------------------------
__global__ void __launch_bounds__(256) k_attn(const __bf16* __restrict__ qkv,
                                              const float*  __restrict__ pos_emb,
                                              __bf16*       __restrict__ attn) {
    __shared__ float pe[225];                                   // 15x15 bias table
    __shared__ __attribute__((aligned(32))) __bf16 pbuf[8][16 * 64]; // per-wave P staging
    if (threadIdx.x < 225) pe[threadIdx.x] = pos_emb[threadIdx.x];
    __syncthreads();

    int wave = threadIdx.x >> 5, lane = threadIdx.x & 31;
    int ln = lane & 15, hi = lane >> 4;
    int unit = blockIdx.x * 8 + wave;          // 16384 = 2048 windows * 8 heads
    int head = unit & 7,  win = unit >> 3;
    int bl = win >> 8, widx = win & 255, wy = widx >> 4, wx = widx & 15;
    int pixBase = bl * 16384 + (wy * 8) * 128 + wx * 8;  // token t -> pixel row pixBase+(t>>3)*128+(t&7)
    int hq = head * 32, hk = 256 + head * 32, hv = 512 + head * 32;

    // K as resident B fragments for S = Q*K^T: B[k][n] = K[token n][c=k]
    bf16x16 kb[4];
#pragma unroll
    for (int j = 0; j < 4; j++) {
        int t = j * 16 + ln;
        size_t row = (size_t)(pixBase + (t >> 3) * 128 + (t & 7));
        kb[j] = *(const bf16x16*)(qkv + row * QKVN + hk + hi * 16);
    }
    // V as resident B fragments for O = P*V: B[k][n] = V[token k][c=n] (gather over tokens)
    bf16x16 vb[2][2];
#pragma unroll
    for (int kf = 0; kf < 2; kf++)
#pragma unroll
        for (int nt = 0; nt < 2; nt++) {
            bf16x16 v;
#pragma unroll
            for (int e = 0; e < 16; e++) {
                int t = kf * 32 + hi * 8 + (e & 7) + ((e & 8) << 1);
                size_t row = (size_t)(pixBase + (t >> 3) * 128 + (t & 7));
                v[e] = qkv[row * QKVN + hv + nt * 16 + ln];
            }
            vb[kf][nt] = v;
        }

    __bf16* pw = &pbuf[wave][0];
    for (int i = 0; i < 4; i++) {            // 16-row tiles of the 64-token window
        // Q A-fragment: row = i*16 + lane%16; K-range split by lane half
        int tq = i * 16 + ln;
        size_t qrow = (size_t)(pixBase + (tq >> 3) * 128 + (tq & 7)) * QKVN + hq;
        union { bf16x16 v; bf16x8 h[2]; } qa;
        qa.h[0] = *(const bf16x8*)(qkv + qrow + hi * 16);
        qa.h[1] = *(const bf16x8*)(qkv + qrow + hi * 16 + 8);

        f32x8 s[4];
#pragma unroll
        for (int j = 0; j < 4; j++) {
            f32x8 z = {};
            s[j] = wmma_bf16(qa.v, kb[j], z);
        }

        // scale + rel-pos bias + row softmax (rows live in 16-lane halves)
        float pv[4][8];
#pragma unroll
        for (int r = 0; r < 8; r++) {
            int irow = i * 16 + r + 8 * hi;                 // query token 0..63
            int iy = irow >> 3, ix = irow & 7;
            float m = -1e30f;
#pragma unroll
            for (int j = 0; j < 4; j++) {
                int jcol = j * 16 + ln;                     // key token 0..63
                int jy = jcol >> 3, jx = jcol & 7;
                float b = pe[(jy - iy + 7) * 15 + (jx - ix + 7)];
                float val = s[j][r] * SCALE + b;
                pv[j][r] = val;
                m = fmaxf(m, val);
            }
            m = fmaxf(m, __shfl_xor(m, 1, 32));
            m = fmaxf(m, __shfl_xor(m, 2, 32));
            m = fmaxf(m, __shfl_xor(m, 4, 32));
            m = fmaxf(m, __shfl_xor(m, 8, 32));
            float sum = 0.f;
#pragma unroll
            for (int j = 0; j < 4; j++) { float e = __expf(pv[j][r] - m); pv[j][r] = e; sum += e; }
            sum += __shfl_xor(sum, 1, 32);
            sum += __shfl_xor(sum, 2, 32);
            sum += __shfl_xor(sum, 4, 32);
            sum += __shfl_xor(sum, 8, 32);
            float inv = 1.0f / sum;
            int lrow = r + 8 * hi;
#pragma unroll
            for (int j = 0; j < 4; j++)
                pw[lrow * 64 + j * 16 + ln] = f2bf(pv[j][r] * inv);
        }
        // LDS is serviced in-order per wave; fence stops compiler reordering
        asm volatile("" ::: "memory");

        // re-read P in A-fragment layout (rows = lane%16, K split by lane half)
        bf16x16 pa[2];
#pragma unroll
        for (int kf = 0; kf < 2; kf++) {
            union { bf16x16 v; bf16x8 h[2]; } u;
            const __bf16* base = pw + ln * 64 + kf * 32 + hi * 8;
            u.h[0] = *(const bf16x8*)(base);
            u.h[1] = *(const bf16x8*)(base + 16);
            pa[kf] = u.v;
        }
        asm volatile("" ::: "memory");

#pragma unroll
        for (int nt = 0; nt < 2; nt++) {
            f32x8 acc = {};
            acc = wmma_bf16(pa[0], vb[0][nt], acc);
            acc = wmma_bf16(pa[1], vb[1][nt], acc);
#pragma unroll
            for (int r = 0; r < 8; r++) {
                int t = i * 16 + r + 8 * hi;
                size_t row = (size_t)(pixBase + (t >> 3) * 128 + (t & 7));
                attn[row * DIMC + head * 32 + nt * 16 + ln] = f2bf(acc[r]);
            }
        }
    }
}

// ---------------------------------------------------------------------------
// Kernel 3: output projection  out(131072x256,f32) = attn(bf16) * WoutT + b_out
// ---------------------------------------------------------------------------
__global__ void __launch_bounds__(256) k_out(const __bf16* __restrict__ attn,
                                             const __bf16* __restrict__ woT,
                                             const float*  __restrict__ bout,
                                             float*        __restrict__ out) {
    int wave = threadIdx.x >> 5, lane = threadIdx.x & 31;
    int t  = blockIdx.x * 8 + wave;          // 131072 tiles total
    int tM = t >> 4, tN = t & 15;
    int ln = lane & 15, hi = lane >> 4;

    const __bf16* arow = attn + (size_t)(tM * 16 + ln) * DIMC;
    const __bf16* brow = woT  + (size_t)(tN * 16 + ln) * DIMC;

    f32x8 acc = {};
#pragma unroll
    for (int k0 = 0; k0 < DIMC; k0 += 32) {
        union { bf16x16 v; bf16x8 h[2]; } a;
        a.h[0] = *(const bf16x8*)(arow + k0 + hi * 16);
        a.h[1] = *(const bf16x8*)(arow + k0 + hi * 16 + 8);
        bf16x16 b = *(const bf16x16*)(brow + k0 + hi * 16);
        acc = wmma_bf16(a.v, b, acc);
    }
    int col = tN * 16 + ln;
    float bias = bout[col];
#pragma unroll
    for (int r = 0; r < 8; r++) {
        int row = tM * 16 + r + 8 * hi;
        out[(size_t)row * DIMC + col] = acc[r] + bias;
    }
}

// ---------------------------------------------------------------------------
extern "C" void kernel_launch(void* const* d_in, const int* in_sizes, int n_in,
                              void* d_out, int out_size, void* d_ws, size_t ws_size,
                              hipStream_t stream) {
    const float* x       = (const float*)d_in[0];  // (2,4,128,128,256)
    const float* w_qkv   = (const float*)d_in[1];  // (256,768)
    const float* pos_emb = (const float*)d_in[2];  // (15,15)
    const float* w_out   = (const float*)d_in[3];  // (256,256)
    const float* b_out   = (const float*)d_in[4];  // (256,)
    float* out = (float*)d_out;

    // workspace layout (bf16 intermediates): ~269 MB total
    //   [qkv 131072x768][attn 131072x256][wqT 768x256][woT 256x256]
    // xb (bf16 copy of x) ALIASES the attn region: it is consumed by k_qkv
    // before k_attn overwrites that region with the attention output.
    char* ws = (char*)d_ws;
    __bf16* qkv  = (__bf16*)ws;                                         // 131072*768
    __bf16* attn = (__bf16*)(ws + (size_t)TOKENS * QKVN * 2);           // 131072*256
    __bf16* xb   = attn;                                                // alias (see above)
    __bf16* wqT  = (__bf16*)(ws + (size_t)TOKENS * QKVN * 2
                                + (size_t)TOKENS * DIMC * 2);           // 768*256
    __bf16* woT  = wqT + 768 * 256;                                     // 256*256

    k_convert_w<<<1024, 256, 0, stream>>>(w_qkv, w_out, wqT, woT);
    k_convert_x<<<32768, 256, 0, stream>>>(x, xb);               // 33.5M elems / 4
    k_qkv     <<<49152, 256, 0, stream>>>(xb, wqT, qkv);         // 8192*48 tiles / 8 waves
    k_attn    <<<2048, 256, 0, stream>>>(qkv, pos_emb, attn);    // 16384 window-heads / 8
    k_out     <<<16384, 256, 0, stream>>>(attn, woT, b_out, out); // 8192*16 tiles / 8
}
